// LMPNN_Encoder_59313498358354
// MI455X (gfx1250) — compile-verified, matrix operand order checked by reading, not run on hardware
//
#include <hip/hip_runtime.h>
#include <stdint.h>
#include <stddef.h>

// Problem constants (match reference)
#define B_DIM   8192
#define N_NODES 9
#define D_DIM   1024
#define H_DIM   4096

typedef __bf16 bf16;
typedef __bf16   v16bf __attribute__((ext_vector_type(16)));
typedef __bf16   v8bf  __attribute__((ext_vector_type(8)));
typedef __bf16   v4bf  __attribute__((ext_vector_type(4)));
typedef float    v8f   __attribute__((ext_vector_type(8)));
typedef float    fx4   __attribute__((ext_vector_type(4)));
typedef uint32_t u32x4 __attribute__((ext_vector_type(4)));
typedef uint32_t u32x8 __attribute__((ext_vector_type(8)));

// ---------------------------------------------------------------------------
// Transpose + fp32->bf16 convert:  in[R][C] fp32  ->  out[C][R] bf16
// Classic 32x32 LDS tile so both global read and write are coalesced.
// ---------------------------------------------------------------------------
__global__ void __launch_bounds__(1024)
transpose_convert_kernel(const float* __restrict__ in, bf16* __restrict__ out,
                         int R, int C)
{
    __shared__ float tile[32][33];
    const int c = blockIdx.x * 32 + threadIdx.x;
    const int r = blockIdx.y * 32 + threadIdx.y;
    tile[threadIdx.y][threadIdx.x] = in[(size_t)r * C + c];
    __syncthreads();
    const int oc  = blockIdx.y * 32 + threadIdx.x;   // index along R
    const int orw = blockIdx.x * 32 + threadIdx.y;   // index along C
    out[(size_t)orw * R + oc] = (bf16)tile[threadIdx.x][threadIdx.y];
}

// ---------------------------------------------------------------------------
// Message aggregation for the 4 output terms (nodes 2,4,6,8), fp32 -> bf16.
//   t=0: 3*(e4-e3)          + 0.1*e2
//   t=1: 3*(e2+e3+e6-e5)    + 0.1*e4
//   t=2: 3*(e4+e5+e8-e7)    + 0.1*e6
//   t=3: 3*(e6+e7)          + 0.1*e8
// One thread per 4 consecutive d (vectorized float4 loads / bf16x4 store).
// ---------------------------------------------------------------------------
__global__ void __launch_bounds__(256)
agg_kernel(const float* __restrict__ ne, bf16* __restrict__ X)
{
    const unsigned perT = B_DIM * (D_DIM / 4);          // vec4 elems per term
    const unsigned idx  = blockIdx.x * blockDim.x + threadIdx.x;
    const unsigned t    = idx / perT;
    const unsigned rem  = idx % perT;
    const unsigned b    = rem / (D_DIM / 4);
    const unsigned d    = (rem % (D_DIM / 4)) * 4;

    const float* base = ne + ((size_t)b * N_NODES) * D_DIM + d;
    #define LN(n) (*(const fx4*)(base + (n) * D_DIM))
    fx4 v;
    if      (t == 0) v = 3.0f * (LN(4) - LN(3))                 + 0.1f * LN(2);
    else if (t == 1) v = 3.0f * (LN(2) + LN(3) + LN(6) - LN(5)) + 0.1f * LN(4);
    else if (t == 2) v = 3.0f * (LN(4) + LN(5) + LN(8) - LN(7)) + 0.1f * LN(6);
    else             v = 3.0f * (LN(6) + LN(7))                 + 0.1f * LN(8);
    #undef LN

    v4bf o;
    o[0] = (bf16)v[0]; o[1] = (bf16)v[1]; o[2] = (bf16)v[2]; o[3] = (bf16)v[3];
    *(v4bf*)(X + (size_t)idx * 4) = o;
}

// ---------------------------------------------------------------------------
// TDM: issue one 2D tile load (tile: 32 bf16 wide x tile_rows) into LDS.
// D# group0 (4 SGPRs) + group1 (8 SGPRs); 2D tensor -> VADDR2/3 omitted.
// LDS padding: 16B pad after every 64B row -> 80B LDS row pitch, which makes
// the per-lane ds_load_b128 fragment fetches bank-conflict-free.
// ---------------------------------------------------------------------------
__device__ __forceinline__ void tdm_load_tile(uint32_t lds_byte_off,
                                              const bf16* gptr,
                                              uint32_t tile_rows,
                                              uint32_t row_stride_elems)
{
    const uint64_t ga = (uint64_t)(uintptr_t)gptr;
    u32x4 g0;
    g0[0] = 1u;                                      // count=1 (valid), user mode
    g0[1] = lds_byte_off;                            // lds_addr
    g0[2] = (uint32_t)ga;                            // global_addr[31:0]
    g0[3] = ((uint32_t)(ga >> 32) & 0x01FFFFFFu)     // global_addr[56:32]
          | (2u << 30);                              // type = 2 ("image")
    u32x8 g1;
    g1[0] = (1u << 16)                               // data_size = 2 bytes
          | (1u << 20)                               // pad_enable
          | (3u << 22)                               // pad_interval: 16 DWORDs (64B)
          | (3u << 25);                              // pad_amount:    4 DWORDs (16B)
    g1[1] = 0u;                                      // tensor_dim0[15:0]=0 (dim0=1<<30)
    g1[2] = 0x4000u;                                 // tensor_dim0[31:16]; dim1[15:0]=0
    g1[3] = 0x4000u | (32u << 16);                   // tensor_dim1[31:16]; tile_dim0=32
    g1[4] = tile_rows & 0xFFFFu;                     // tile_dim1; tile_dim2=0
    g1[5] = row_stride_elems;                        // tensor_dim0_stride[31:0]
    g1[6] = 0u;                                      // stride0[47:32]=0; stride1 lo=0
    g1[7] = 0u;
    asm volatile("tensor_load_to_lds %0, %1" :: "s"(g0), "s"(g1) : "memory");
}

// ---------------------------------------------------------------------------
// bf16 WMMA GEMM:  C[M][N] = A[M][K] * Bt[N][K]^T + bias[N]   (fp32 accum)
// Block: 256 threads = 8 waves, block tile 128(M) x 256(N), K-step 32.
// Wave tile 64x64 = 4x4 v_wmma_f32_16x16x32_bf16 accumulators.
// A/B tiles staged to LDS by the Tensor Data Mover, double buffered,
// synchronized with s_wait_tensorcnt + workgroup barrier.
// ---------------------------------------------------------------------------
template<bool RELU, bool OUT_BF16>
__global__ void __launch_bounds__(256)
gemm_wmma_bf16(const bf16* __restrict__ A, const bf16* __restrict__ Bt,
               const float* __restrict__ bias, void* __restrict__ Cout,
               int N, int K)
{
    extern __shared__ bf16 smem[];
    constexpr int BM = 128, BN = 256, BK = 32;
    constexpr int ROWB = 40;                         // LDS row pitch in elems (64B+16B pad)
    const uint32_t A_OFF[2] = {0u, 10240u};          // byte offsets of A buffers
    const uint32_t B_OFF[2] = {20480u, 40960u};      // byte offsets of B buffers

    const int lane    = threadIdx.x & 31;
    const int wid     = threadIdx.x >> 5;
    const int lane_lo = lane & 15;                   // M (A) / N (B,C,D) index
    const int lane_hi = lane >> 4;                   // K-half (A,B) / M-half (C,D)
    const int wm      = (wid >> 2) << 6;             // wave M origin: 0 or 64
    const int wn      = (wid & 3) << 6;              // wave N origin: 0..192

    const bf16* Ablk = A  + (size_t)blockIdx.y * BM * K;
    const bf16* Bblk = Bt + (size_t)blockIdx.x * BN * K;

    v8f acc[4][4];
#pragma unroll
    for (int i = 0; i < 4; ++i)
#pragma unroll
        for (int j = 0; j < 4; ++j)
            acc[i][j] = (v8f){0.f, 0.f, 0.f, 0.f, 0.f, 0.f, 0.f, 0.f};

    const int nsteps = K / BK;

    // Prologue: wave 0 kicks off the first pair of tile loads.
    if (threadIdx.x == 0) {
        tdm_load_tile(A_OFF[0], Ablk, BM, (uint32_t)K);
        tdm_load_tile(B_OFF[0], Bblk, BN, (uint32_t)K);
    }

    for (int s = 0; s < nsteps; ++s) {
        // Wave 0's pending TDM loads for buffer s%2 must land; other waves
        // have TENSORcnt==0 and pass straight to the barrier.
        __builtin_amdgcn_s_wait_tensorcnt(0);
        __syncthreads();   // everyone: buffer s%2 ready, buffer (s+1)%2 free

        if ((threadIdx.x == 0) && (s + 1 < nsteps)) {
            const int nb = (s + 1) & 1;
            tdm_load_tile(A_OFF[nb], Ablk + (size_t)(s + 1) * BK, BM, (uint32_t)K);
            tdm_load_tile(B_OFF[nb], Bblk + (size_t)(s + 1) * BK, BN, (uint32_t)K);
        }

        const int cb = s & 1;
        const bf16* As = smem + (A_OFF[cb] >> 1);
        const bf16* Bs = smem + (B_OFF[cb] >> 1);

        // Fragment fetches per ISA layout: lane_lo = row, K runs
        // [8*lane_hi .. +7] and [16+8*lane_hi .. +7] -> two b128 LDS reads.
        v16bf afr[4], bfr[4];
#pragma unroll
        for (int i = 0; i < 4; ++i) {
            const bf16* p = As + (wm + 16 * i + lane_lo) * ROWB + 8 * lane_hi;
            v8bf lo = *(const v8bf*)(p);
            v8bf hi = *(const v8bf*)(p + 16);
            afr[i] = __builtin_shufflevector(lo, hi,
                     0,1,2,3,4,5,6,7,8,9,10,11,12,13,14,15);
        }
#pragma unroll
        for (int j = 0; j < 4; ++j) {
            const bf16* p = Bs + (wn + 16 * j + lane_lo) * ROWB + 8 * lane_hi;
            v8bf lo = *(const v8bf*)(p);
            v8bf hi = *(const v8bf*)(p + 16);
            bfr[j] = __builtin_shufflevector(lo, hi,
                     0,1,2,3,4,5,6,7,8,9,10,11,12,13,14,15);
        }

#pragma unroll
        for (int i = 0; i < 4; ++i)
#pragma unroll
            for (int j = 0; j < 4; ++j)
                acc[i][j] = __builtin_amdgcn_wmma_f32_16x16x32_bf16(
                    false, afr[i], false, bfr[j], (short)0, acc[i][j],
                    false, false);
    }

    // Epilogue: C/D layout -> lane_lo = N, M = 8*lane_hi + vgpr index.
    const int m_base = blockIdx.y * BM + wm + 8 * lane_hi;
    const int n_base = blockIdx.x * BN + wn + lane_lo;
#pragma unroll
    for (int j = 0; j < 4; ++j) {
        const int n   = n_base + 16 * j;
        const float bj = bias[n];
#pragma unroll
        for (int i = 0; i < 4; ++i) {
#pragma unroll
            for (int r = 0; r < 8; ++r) {
                float v = acc[i][j][r] + bj;
                if (RELU) v = (v > 0.f) ? v : 0.f;
                const size_t off = (size_t)(m_base + 16 * i + r) * N + n;
                if (OUT_BF16) ((bf16*)Cout)[off] = (bf16)v;
                else          ((float*)Cout)[off] = v;
            }
        }
    }
}

// ---------------------------------------------------------------------------
// Host-side orchestration.
// Workspace layout (144 MB):
//   [0,   8MB)  W1^T bf16  [H][D]
//   [8,  16MB)  W2^T bf16  [D][H]
//   [16, 80MB)  X    bf16  [4][B][D]   aggregated MLP inputs
//   [80,144MB)  Hidden bf16 [B][H]     reused across the 4 terms
// ---------------------------------------------------------------------------
extern "C" void kernel_launch(void* const* d_in, const int* in_sizes, int n_in,
                              void* d_out, int out_size, void* d_ws, size_t ws_size,
                              hipStream_t stream)
{
    (void)in_sizes; (void)n_in; (void)out_size; (void)ws_size;

    const float* ne = (const float*)d_in[0];   // (B, 9, D)
    const float* W1 = (const float*)d_in[1];   // (D, H)
    const float* b1 = (const float*)d_in[2];   // (H)
    const float* W2 = (const float*)d_in[3];   // (H, D)
    const float* b2 = (const float*)d_in[4];   // (D)
    float* out = (float*)d_out;                // (4, B, D)

    char* ws = (char*)d_ws;
    bf16* W1T    = (bf16*)(ws);
    bf16* W2T    = (bf16*)(ws + ((size_t)8  << 20));
    bf16* X      = (bf16*)(ws + ((size_t)16 << 20));
    bf16* Hidden = (bf16*)(ws + ((size_t)80 << 20));

    // Weight transpose+convert (fp32 -> bf16, [K][N] -> [N][K]).
    {
        dim3 tb(32, 32);
        transpose_convert_kernel<<<dim3(H_DIM / 32, D_DIM / 32), tb, 0, stream>>>(
            W1, W1T, D_DIM, H_DIM);
        transpose_convert_kernel<<<dim3(D_DIM / 32, H_DIM / 32), tb, 0, stream>>>(
            W2, W2T, H_DIM, D_DIM);
    }

    // Aggregate messages for all 4 terms into X (bf16).
    {
        const unsigned total_v4 = 4u * B_DIM * (D_DIM / 4);
        agg_kernel<<<total_v4 / 256, 256, 0, stream>>>(ne, X);
    }

    // Per term: Hidden = relu(X_t @ W1 + b1) ; out_t = Hidden @ W2 + b2.
    const size_t termA = (size_t)B_DIM * D_DIM;
    const unsigned shmem = 61440;  // 2*(128*40 + 256*40)*2 bytes (padded tiles)
    for (int t = 0; t < 4; ++t) {
        gemm_wmma_bf16<true, true>
            <<<dim3(H_DIM / 256, B_DIM / 128), 256, shmem, stream>>>(
                X + (size_t)t * termA, W1T, b1, Hidden, H_DIM, D_DIM);
        gemm_wmma_bf16<false, false>
            <<<dim3(D_DIM / 256, B_DIM / 128), 256, shmem, stream>>>(
                Hidden, W2T, b2, out + (size_t)t * termA, D_DIM, H_DIM);
    }
}
